// ESM2ContactPredictor_45956150067395
// MI455X (gfx1250) — matrix operand order repaired and checked by reading, not compiled, not run on hardware
//
#include <hip/hip_runtime.h>

// ESM2 contact head: out[b,i,j] = sigmoid(x[b,i] @ W @ x[b,j] + bias)
// B=8, L=2050 (padded to 2080), D=320.
//
// Pipeline (all bf16x3 split-precision WMMA, fp32 accumulate):
//   k_split_x : x fp32 -> (x_hi, x_lo) bf16, zero-padded rows to L_PAD
//   k_split_w : W fp32 -> transposed (wt_hi, wt_lo) bf16
//   k_gemm1   : y = x @ W  via v_wmma_f32_16x16x32_bf16, epilogue re-splits y
//   k_gemm2   : logits = y @ x^T (2x2 tile blocking), + bias, sigmoid, fp32 out
//
// Workspace: 4 * (8*2080*320) + 2 * (320*320) bf16 elems ~= 43 MB.

#define L_SEQ  2050
#define L_PAD  2080      // multiple of 32
#define DIM    320       // multiple of 32
#define NBATCH 8

typedef __attribute__((ext_vector_type(16))) __bf16        v16bf;
typedef __attribute__((ext_vector_type(8)))  float         v8f;
typedef __attribute__((ext_vector_type(4)))  unsigned int  v4u;

struct Q2 { v4u a, b; };   // 32 bytes == one v16bf fragment

static __device__ __forceinline__ unsigned short f2bf_rn(float f) {
    unsigned u = __float_as_uint(f);
    u += 0x7FFFu + ((u >> 16) & 1u);          // round-to-nearest-even
    return (unsigned short)(u >> 16);
}
static __device__ __forceinline__ float bf2f(unsigned short h) {
    return __uint_as_float(((unsigned)h) << 16);
}
static __device__ __forceinline__ float sigmoidf(float t) {
    return 1.0f / (1.0f + __expf(-t));
}

static __device__ __forceinline__ v16bf frag_pair(const unsigned short* p0,
                                                  const unsigned short* p1) {
    Q2 q;
    q.a = *(const v4u*)p0;   // 8 bf16, 16B aligned
    q.b = *(const v4u*)p1;
    return __builtin_bit_cast(v16bf, q);
}

// A fragment: 16x32 bf16 tile at (rowBase, k0) of row-major [row][k], ld = DIM.
// lane<16 -> row=rowBase+lane, K in {k0+0..7, k0+16..23}; lane>=16 -> K +8.
static __device__ __forceinline__ v16bf load_fragA(const unsigned short* base,
                                                   int rowBase, int k0, int lane) {
    int lr = lane & 15, half = lane >> 4;
    const unsigned short* r = base + (size_t)(rowBase + lr) * DIM;
    return frag_pair(r + k0 + half * 8, r + k0 + 16 + half * 8);
}

// B fragment: 32x16 bf16 tile; B[k][n] = base[(colBase+n)*DIM + k] (K contiguous
// per column). lane<16 -> col=colBase+lane, K in {k0+0..7, k0+16..23}; lane>=16 -> K +8.
static __device__ __forceinline__ v16bf load_fragB(const unsigned short* base,
                                                   int colBase, int k0, int lane) {
    int lr = lane & 15, half = lane >> 4;
    const unsigned short* r = base + (size_t)(colBase + lr) * DIM + k0 + half * 8;
    return frag_pair(r, r + 16);
}

#define WMMA_BF16(A, Bm, C) \
    __builtin_amdgcn_wmma_f32_16x16x32_bf16(false, (A), false, (Bm), (short)0, (C), false, false)

// ---------------------------------------------------------------- split x
__global__ void k_split_x(const float* __restrict__ x,
                          unsigned short* __restrict__ xh,
                          unsigned short* __restrict__ xl) {
    size_t idx = (size_t)blockIdx.x * blockDim.x + threadIdx.x;
    const size_t total = (size_t)NBATCH * L_PAD * DIM;
    if (idx >= total) return;
    int d   = (int)(idx % DIM);
    int row = (int)((idx / DIM) % L_PAD);
    int b   = (int)(idx / ((size_t)DIM * L_PAD));
    float v = 0.0f;
    if (row < L_SEQ) v = x[((size_t)b * L_SEQ + row) * DIM + d];
    unsigned short hi = f2bf_rn(v);
    xh[idx] = hi;
    xl[idx] = f2bf_rn(v - bf2f(hi));
}

// ------------------------------------------------------ split + transpose W
__global__ void k_split_w(const float* __restrict__ W,
                          unsigned short* __restrict__ wth,
                          unsigned short* __restrict__ wtl) {
    int idx = blockIdx.x * blockDim.x + threadIdx.x;
    if (idx >= DIM * DIM) return;
    int e = idx % DIM, d = idx / DIM;       // W[d][e]
    float v = W[idx];
    unsigned short hi = f2bf_rn(v);
    wth[e * DIM + d] = hi;                  // Wt[e][d]
    wtl[e * DIM + d] = f2bf_rn(v - bf2f(hi));
}

// ------------------------------------------------- GEMM1: y = x @ W (bf16x3)
__global__ void k_gemm1(const unsigned short* __restrict__ xh,
                        const unsigned short* __restrict__ xl,
                        const unsigned short* __restrict__ wth,
                        const unsigned short* __restrict__ wtl,
                        unsigned short* __restrict__ yh,
                        unsigned short* __restrict__ yl) {
    const int TI = L_PAD / 16, TE = DIM / 16;
    int wave = blockIdx.x * (blockDim.x >> 5) + (threadIdx.x >> 5);
    int lane = threadIdx.x & 31;
    if (wave >= NBATCH * TI * TE) return;   // wave-uniform: EXEC all-1 below
    int b   = wave / (TI * TE);
    int rem = wave % (TI * TE);
    int iBase = (rem / TE) * 16;
    int eBase = (rem % TE) * 16;

    const unsigned short* ah = xh + (size_t)b * L_PAD * DIM;
    const unsigned short* al = xl + (size_t)b * L_PAD * DIM;

    v8f acc = {0.f, 0.f, 0.f, 0.f, 0.f, 0.f, 0.f, 0.f};
    for (int k0 = 0; k0 < DIM; k0 += 32) {
        v16bf aH = load_fragA(ah,  iBase, k0, lane);
        v16bf aL = load_fragA(al,  iBase, k0, lane);
        v16bf bH = load_fragB(wth, eBase, k0, lane);
        v16bf bL = load_fragB(wtl, eBase, k0, lane);
        acc = WMMA_BF16(aH, bH, acc);
        acc = WMMA_BF16(aH, bL, acc);
        acc = WMMA_BF16(aL, bH, acc);
    }

    int lr = lane & 15, half = lane >> 4;
    unsigned short* oyh = yh + (size_t)b * L_PAD * DIM;
    unsigned short* oyl = yl + (size_t)b * L_PAD * DIM;
#pragma unroll
    for (int v = 0; v < 8; ++v) {
        int i = iBase + v + 8 * half;       // C/D layout: M = vgpr + 8*half
        int e = eBase + lr;                 // N = lane % 16
        float f = acc[v];
        unsigned short hi = f2bf_rn(f);
        oyh[(size_t)i * DIM + e] = hi;
        oyl[(size_t)i * DIM + e] = f2bf_rn(f - bf2f(hi));
    }
}

// ---------------------- GEMM2: sigmoid(y @ x^T + bias), 2x2 WMMA tile block
__global__ void k_gemm2(const unsigned short* __restrict__ yh,
                        const unsigned short* __restrict__ yl,
                        const unsigned short* __restrict__ xh,
                        const unsigned short* __restrict__ xl,
                        const float* __restrict__ bias,
                        float* __restrict__ out) {
    const int T = L_PAD / 32;               // 65 macro-tiles per dim
    int wave = blockIdx.x * (blockDim.x >> 5) + (threadIdx.x >> 5);
    int lane = threadIdx.x & 31;
    if (wave >= NBATCH * T * T) return;     // wave-uniform guard
    int b   = wave / (T * T);
    int rem = wave % (T * T);
    int ti  = rem / T, tj = rem % T;
    int m0 = ti * 32, m1 = m0 + 16;
    int n0 = tj * 32, n1 = n0 + 16;

    const unsigned short* Ah = yh + (size_t)b * L_PAD * DIM;
    const unsigned short* Al = yl + (size_t)b * L_PAD * DIM;
    const unsigned short* Bh = xh + (size_t)b * L_PAD * DIM;   // x^T: col j -> row j
    const unsigned short* Bl = xl + (size_t)b * L_PAD * DIM;

    v8f acc00 = {0.f,0.f,0.f,0.f,0.f,0.f,0.f,0.f};
    v8f acc01 = acc00, acc10 = acc00, acc11 = acc00;

    for (int k0 = 0; k0 < DIM; k0 += 32) {
        v16bf aH0 = load_fragA(Ah, m0, k0, lane);
        v16bf aL0 = load_fragA(Al, m0, k0, lane);
        v16bf aH1 = load_fragA(Ah, m1, k0, lane);
        v16bf aL1 = load_fragA(Al, m1, k0, lane);
        v16bf bH0 = load_fragB(Bh, n0, k0, lane);
        v16bf bL0 = load_fragB(Bl, n0, k0, lane);
        v16bf bH1 = load_fragB(Bh, n1, k0, lane);
        v16bf bL1 = load_fragB(Bl, n1, k0, lane);

        acc00 = WMMA_BF16(aH0, bH0, acc00);
        acc00 = WMMA_BF16(aH0, bL0, acc00);
        acc00 = WMMA_BF16(aL0, bH0, acc00);

        acc01 = WMMA_BF16(aH0, bH1, acc01);
        acc01 = WMMA_BF16(aH0, bL1, acc01);
        acc01 = WMMA_BF16(aL0, bH1, acc01);

        acc10 = WMMA_BF16(aH1, bH0, acc10);
        acc10 = WMMA_BF16(aH1, bL0, acc10);
        acc10 = WMMA_BF16(aL1, bH0, acc10);

        acc11 = WMMA_BF16(aH1, bH1, acc11);
        acc11 = WMMA_BF16(aH1, bL1, acc11);
        acc11 = WMMA_BF16(aL1, bH1, acc11);
    }

    float bv = bias[0];
    int lr = lane & 15, half = lane >> 4;
    float* ob = out + (size_t)b * L_SEQ * L_SEQ;

#define STORE_TILE(ACC, MI, NJ)                                          \
    _Pragma("unroll")                                                    \
    for (int v = 0; v < 8; ++v) {                                        \
        int i = (MI) + v + 8 * half;                                     \
        int j = (NJ) + lr;                                               \
        if (i < L_SEQ && j < L_SEQ)                                      \
            ob[(size_t)i * L_SEQ + j] = sigmoidf((ACC)[v] + bv);         \
    }
    STORE_TILE(acc00, m0, n0)
    STORE_TILE(acc01, m0, n1)
    STORE_TILE(acc10, m1, n0)
    STORE_TILE(acc11, m1, n1)
#undef STORE_TILE
}

// --------------------------------------------------------------- launcher
extern "C" void kernel_launch(void* const* d_in, const int* in_sizes, int n_in,
                              void* d_out, int out_size, void* d_ws, size_t ws_size,
                              hipStream_t stream) {
    (void)in_sizes; (void)n_in; (void)out_size; (void)ws_size;
    const float* x    = (const float*)d_in[0];   // (8, 2050, 320)
    const float* W    = (const float*)d_in[1];   // (320, 320)
    const float* bias = (const float*)d_in[2];   // (1,)
    float* out = (float*)d_out;                  // (8, 2050, 2050) fp32

    const size_t xsz = (size_t)NBATCH * L_PAD * DIM;     // elems per split array
    unsigned short* xh  = (unsigned short*)d_ws;
    unsigned short* xl  = xh + xsz;
    unsigned short* yh  = xl + xsz;
    unsigned short* yl  = yh + xsz;
    unsigned short* wth = yl + xsz;
    unsigned short* wtl = wth + (size_t)DIM * DIM;       // total ~43 MB

    {   // split/pad x
        int threads = 256;
        int blocks = (int)((xsz + threads - 1) / threads);
        k_split_x<<<blocks, threads, 0, stream>>>(x, xh, xl);
    }
    {   // split + transpose W
        k_split_w<<<(DIM * DIM + 255) / 256, 256, 0, stream>>>(W, wth, wtl);
    }
    {   // y = x @ W
        int waves  = NBATCH * (L_PAD / 16) * (DIM / 16);
        int blocks = (waves + 7) / 8;
        k_gemm1<<<blocks, 256, 0, stream>>>(xh, xl, wth, wtl, yh, yl);
    }
    {   // out = sigmoid(y @ x^T + bias)
        int T = L_PAD / 32;
        int waves  = NBATCH * T * T;
        int blocks = (waves + 7) / 8;
        k_gemm2<<<blocks, 256, 0, stream>>>(yh, yl, xh, xl, bias, out);
    }
}
